// Transformer_26731876450517
// MI455X (gfx1250) — matrix operand order repaired and checked by reading, not compiled, not run on hardware
//
#include <hip/hip_runtime.h>
#include <hip/hip_bf16.h>
#include <math.h>

// ---------------------------------------------------------------------------
// Transformer forward (encoder-decoder) for MI455X / gfx1250.
// GEMMs: v_wmma_f32_16x16x32_bf16, fp32 accumulate, bf16 staged through LDS.
// Q/K/V are produced in bf16 by the projection GEMMs; fused flash-attention
// stages K tiles with the Tensor Data Mover (tensor_load_to_lds +
// s_wait_tensorcnt) and runs QK^T / PV on WMMA.
// ---------------------------------------------------------------------------

#define DEV __device__ __forceinline__

typedef __attribute__((ext_vector_type(16))) __bf16          v16bf;
typedef __attribute__((ext_vector_type(16))) unsigned short  v16u;
typedef __attribute__((ext_vector_type(8)))  unsigned short  v8u;
typedef __attribute__((ext_vector_type(8)))  float           v8f;
typedef __attribute__((ext_vector_type(4)))  unsigned int    v4u;
typedef __attribute__((ext_vector_type(8)))  int             v8i;
typedef __attribute__((ext_vector_type(4)))  int             v4i;

#if __has_builtin(__builtin_amdgcn_tensor_load_to_lds) && \
    __has_builtin(__builtin_amdgcn_s_wait_tensorcnt)
#define USE_TDM 1
#else
#define USE_TDM 0
#endif

constexpr int Bc = 4, Sc = 512, Dc = 512, Hc = 8, Fc = 2048, Lc = 6, Vc = 32000;
constexpr int DKc = 64;
constexpr int MR = Bc * Sc;  // 2048 rows (== B*T as well)

DEV unsigned short f2bf(float f) {
  return __builtin_bit_cast(unsigned short, (__bf16)f);  // hw cvt, RNE
}

#if __has_builtin(__builtin_amdgcn_cvt_pk_bf16_f32)
DEV unsigned int f2bf2(float a, float b) {
  return __builtin_bit_cast(unsigned int, __builtin_amdgcn_cvt_pk_bf16_f32(a, b));
}
#else
DEV unsigned int f2bf2(float a, float b) {
  return (unsigned int)f2bf(a) | ((unsigned int)f2bf(b) << 16);
}
#endif

DEV v16bf cat_bf(v8u lo, v8u hi) {
  return __builtin_bit_cast(
      v16bf, __builtin_shufflevector(lo, hi, 0, 1, 2, 3, 4, 5, 6, 7, 8, 9, 10,
                                     11, 12, 13, 14, 15));
}

DEV v8f wmma_bf16(v16bf a, v16bf b, v8f c) {
  return __builtin_amdgcn_wmma_f32_16x16x32_bf16(false, a, false, b, (short)0,
                                                 c, false, false);
}

// ---------------------------------------------------------------------------
// GEMM: C = act(A[M,K] @ W + bias).  BT=0: W is [K][N]; BT=1: W is [N][K].
// MODE: 0 = f32 out, 1 = f32 ReLU out, 2 = bf16 out (for Q/K/V projections).
// 128x128 block tile, 8 waves, each wave a 32x64 sub-tile (2x4 WMMA tiles).
// ---------------------------------------------------------------------------
template <int MODE, int BT>
__launch_bounds__(256) __global__
void gemm_bias_kernel(const float* __restrict__ A, const float* __restrict__ W,
                      const float* __restrict__ bias, void* __restrict__ Cv,
                      int M, int N, int K) {
  constexpr int BKP = 40;  // 32 + 8 pad (16B-aligned runs, spread banks)
  __shared__ alignas(64) unsigned short As[128][BKP];  // [m][k] bf16
  __shared__ alignas(64) unsigned short Bs[128][BKP];  // [n][k] bf16

  const int tid = threadIdx.x;
  const int lane = tid & 31, wid = tid >> 5;
  const int half = lane >> 4, l16 = lane & 15;
  const int mBase = blockIdx.y * 128, nBase = blockIdx.x * 128;
  const int wm = (wid >> 1) * 32;  // wave row offset: 0/32/64/96
  const int wn = (wid & 1) * 64;   // wave col offset: 0/64

  v8f acc[2][4] = {};

  for (int kt = 0; kt < K; kt += 32) {
    // ---- stage A tile (128x32) as bf16, [m][k] ----
    {
      const int kc = (tid & 7) * 4;
      for (int i = 0; i < 4; ++i) {
        const int row = (tid >> 3) + i * 32;
        const float4 f = *reinterpret_cast<const float4*>(
            &A[(size_t)(mBase + row) * K + kt + kc]);
        *reinterpret_cast<unsigned int*>(&As[row][kc]) = f2bf2(f.x, f.y);
        *reinterpret_cast<unsigned int*>(&As[row][kc + 2]) = f2bf2(f.z, f.w);
      }
      if (kt + 32 < K)
        __builtin_prefetch(&A[(size_t)(mBase + (tid >> 3)) * K + kt + 32 + kc],
                           0, 1);
    }
    // ---- stage B tile as bf16, transposed to [n][k] ----
    if (BT) {
      const int kc = (tid & 7) * 4;
      for (int i = 0; i < 4; ++i) {
        const int col = (tid >> 3) + i * 32;
        const float4 f = *reinterpret_cast<const float4*>(
            &W[(size_t)(nBase + col) * K + kt + kc]);
        *reinterpret_cast<unsigned int*>(&Bs[col][kc]) = f2bf2(f.x, f.y);
        *reinterpret_cast<unsigned int*>(&Bs[col][kc + 2]) = f2bf2(f.z, f.w);
      }
    } else {
      const int nc = (tid & 31) * 4;
      for (int i = 0; i < 4; ++i) {
        const int k = (tid >> 5) + i * 8;
        const float4 f = *reinterpret_cast<const float4*>(
            &W[(size_t)(kt + k) * N + nBase + nc]);
        Bs[nc + 0][k] = f2bf(f.x);
        Bs[nc + 1][k] = f2bf(f.y);
        Bs[nc + 2][k] = f2bf(f.z);
        Bs[nc + 3][k] = f2bf(f.w);
      }
      if (kt + 32 < K)
        __builtin_prefetch(&W[(size_t)(kt + 32 + (tid >> 5)) * N + nBase + nc],
                           0, 1);
    }
    __syncthreads();

    // ---- fragments per ISA 16-bit layouts ----
    v16bf af[2], bfv[4];
    for (int mi = 0; mi < 2; ++mi) {
      const int row = wm + mi * 16 + l16;
      const int kh = half * 8;  // lanes 0-15: K 0..7/16..23; 16-31: +8
      const v8u lo = *reinterpret_cast<const v8u*>(&As[row][kh]);
      const v8u hi = *reinterpret_cast<const v8u*>(&As[row][kh + 16]);
      af[mi] = cat_bf(lo, hi);
    }
    for (int ni = 0; ni < 4; ++ni) {
      const int col = wn + ni * 16 + l16;
      const int kb = half * 8;  // lanes 0-15: K 0..7/16..23; 16-31: +8
      const v8u lo = *reinterpret_cast<const v8u*>(&Bs[col][kb]);
      const v8u hi = *reinterpret_cast<const v8u*>(&Bs[col][kb + 16]);
      bfv[ni] = cat_bf(lo, hi);
    }
    for (int mi = 0; mi < 2; ++mi)
      for (int ni = 0; ni < 4; ++ni)
        acc[mi][ni] = wmma_bf16(af[mi], bfv[ni], acc[mi][ni]);
    __syncthreads();
  }

  // ---- epilogue ----
  for (int mi = 0; mi < 2; ++mi) {
    for (int ni = 0; ni < 4; ++ni) {
      const int n = nBase + wn + ni * 16 + l16;
      const float bv = bias ? bias[n] : 0.f;
      for (int r = 0; r < 8; ++r) {
        const int m = mBase + wm + mi * 16 + r + half * 8;
        float vv = acc[mi][ni][r] + bv;
        if constexpr (MODE == 1) vv = fmaxf(vv, 0.f);
        if constexpr (MODE == 2)
          ((unsigned short*)Cv)[(size_t)m * N + n] = f2bf(vv);
        else
          ((float*)Cv)[(size_t)m * N + n] = vv;
      }
    }
  }
}

// ---------------------------------------------------------------------------
// Fused flash attention on bf16 Q/K/V. One block per (b,h, 128-query tile); 8 waves x 16
// query rows. K tiles staged via the Tensor Data Mover (with TDM LDS padding
// to a 144B row pitch); V staged transposed; P transposed through per-wave
// LDS with an explicit s_wait_dscnt. Mask semantics match the reference
// (scale, then masked positions = -10000).
// ---------------------------------------------------------------------------
__launch_bounds__(256) __global__
void attention_kernel(const unsigned short* __restrict__ Qh,
                      const unsigned short* __restrict__ Kh,
                      const unsigned short* __restrict__ Vh,
                      const int* __restrict__ kvseq, float* __restrict__ O,
                      int SKV, int causal) {
  constexpr int KP = 72;  // Ks row: 64 dk + 8 pad (matches TDM pad 4 DWORDs)
  constexpr int VP = 40;  // Vs/Pb row: 32 keys + 8 pad
  __shared__ alignas(64) unsigned short Ks[32][KP];     // [key][dk]
  __shared__ alignas(64) unsigned short Vs[64][VP];     // [dk][key]
  __shared__ alignas(64) unsigned short Pb[8][16][VP];  // per-wave P tile

  const int tid = threadIdx.x;
  const int lane = tid & 31, wid = tid >> 5;
  const int half = lane >> 4, l16 = lane & 15;
  const int b = blockIdx.x / Hc, h = blockIdx.x % Hc;
  const int q0 = blockIdx.y * 128 + wid * 16;

  // Q fragments: 16x64 bf16 -> two 16x32 A-frags (direct global loads)
  v16bf qa[2];
  {
    const unsigned short* qrow = &Qh[((size_t)(b * Sc + q0 + l16)) * Dc + h * DKc];
    for (int kf = 0; kf < 2; ++kf) {
      const int d0 = kf * 32 + half * 8;
      const v8u lo = *reinterpret_cast<const v8u*>(qrow + d0);
      const v8u hi = *reinterpret_cast<const v8u*>(qrow + d0 + 16);
      qa[kf] = cat_bf(lo, hi);
    }
  }

  float mrow[8], lrow[8];
  v8f oacc[4] = {};
  for (int r = 0; r < 8; ++r) { mrow[r] = -1e30f; lrow[r] = 0.f; }

#if USE_TDM
  const unsigned ks_lds = (unsigned)(size_t)(void*)&Ks[0][0];
  // D# group 1 (constant): data_size=2B, pad_enable, pad_interval=32 DWORDs,
  // pad_amount=4 DWORDs; tensor 64x32, tile 64x32, dim0 stride = D elements.
  const v8i g1 = {(int)0x07110000,       // mask=0|size=1|pad_en|intv=4|amt=3
                  (int)0x00400000,       // tensor_dim0 = 64 (low 16 in hi half)
                  (int)0x00200000,       // tensor_dim1 = 32
                  (int)0x00400000,       // tile_dim0 = 64
                  32,                    // tile_dim1 = 32, tile_dim2 = 0
                  512,                   // tensor_dim0_stride = D
                  0, 0};
  const v4i gz4 = {};
  const v8i gz8 = {};
#endif

  for (int kv0 = 0; kv0 < SKV; kv0 += 32) {
    // ---- K tile staging: TDM async DMA (wave 0 issues) ----
#if USE_TDM
    if (wid == 0) {
      const unsigned long long ga = (unsigned long long)(size_t)(const void*)(
          Kh + ((size_t)(b * SKV + kv0)) * Dc + h * DKc);
      v4u g0;
      g0[0] = 1u;  // count=1 descriptor
      g0[1] = ks_lds;
      g0[2] = (unsigned)ga;
      g0[3] = (unsigned)(ga >> 32) | 0x80000000u;  // type=2 ("image")
      __builtin_amdgcn_tensor_load_to_lds(g0, g1, gz4, gz4, gz8, 0);
    }
#else
    {
      const int key = tid >> 3;
      const int d0 = (tid & 7) * 8;
      const v8u kk = *reinterpret_cast<const v8u*>(
          &Kh[((size_t)(b * SKV + kv0 + key)) * Dc + h * DKc + d0]);
      *reinterpret_cast<v8u*>(&Ks[key][d0]) = kk;
    }
#endif
    // ---- V tile staging (transposed to [dk][key]) ----
    {
      const int key = tid >> 3;
      const int d0 = (tid & 7) * 8;
      const v8u vv = *reinterpret_cast<const v8u*>(
          &Vh[((size_t)(b * SKV + kv0 + key)) * Dc + h * DKc + d0]);
      for (int j = 0; j < 8; ++j) Vs[d0 + j][key] = vv[j];
    }
#if USE_TDM
    if (wid == 0) __builtin_amdgcn_s_wait_tensorcnt(0);
#endif
    __syncthreads();

    // ---- scores: two 16x16 C tiles over the 32-key step ----
    v8f sc[2];
    for (int nt = 0; nt < 2; ++nt) {
      const int keyl = nt * 16 + l16;
      v8f c = {};
      for (int kk = 0; kk < 2; ++kk) {
        const int o = kk * 32 + half * 8;
        const v8u lo = *reinterpret_cast<const v8u*>(&Ks[keyl][o]);
        const v8u hi = *reinterpret_cast<const v8u*>(&Ks[keyl][o + 16]);
        c = wmma_bf16(qa[kk], cat_bf(lo, hi), c);
      }
      sc[nt] = c;
    }

    // ---- online softmax update ----
    const int kg0 = kv0 + l16;
    const int kg1 = kv0 + 16 + l16;
    const int valid0 = kvseq[b * SKV + kg0] != 0;
    const int valid1 = kvseq[b * SKV + kg1] != 0;
    for (int r = 0; r < 8; ++r) {
      const int qg = q0 + r + half * 8;
      float s0 = sc[0][r] * 0.125f;  // 1/sqrt(64)
      float s1 = sc[1][r] * 0.125f;
      if (!valid0 || (causal && kg0 > qg)) s0 = -10000.f;
      if (!valid1 || (causal && kg1 > qg)) s1 = -10000.f;
      float mx = fmaxf(s0, s1);
      for (int m = 1; m <= 8; m <<= 1) mx = fmaxf(mx, __shfl_xor(mx, m));
      const float mnew = fmaxf(mrow[r], mx);
      const float cf = __expf(mrow[r] - mnew);
      const float p0 = __expf(s0 - mnew);
      const float p1 = __expf(s1 - mnew);
      float sum = p0 + p1;
      for (int m = 1; m <= 8; m <<= 1) sum += __shfl_xor(sum, m);
      lrow[r] = lrow[r] * cf + sum;
      mrow[r] = mnew;
      for (int dt = 0; dt < 4; ++dt) oacc[dt][r] *= cf;
      Pb[wid][r + half * 8][l16] = f2bf(p0);
      Pb[wid][r + half * 8][16 + l16] = f2bf(p1);
    }
    // per-wave LDS store->load hazard: wait on DScnt explicitly
    asm volatile("s_wait_dscnt 0" ::: "memory");

    // ---- P (16x32) A-frag, then O += P*V (4 dk tiles) ----
    v16bf pa;
    {
      const int kh = half * 8;
      const v8u lo = *reinterpret_cast<const v8u*>(&Pb[wid][l16][kh]);
      const v8u hi = *reinterpret_cast<const v8u*>(&Pb[wid][l16][kh + 16]);
      pa = cat_bf(lo, hi);
    }
    for (int dt = 0; dt < 4; ++dt) {
      const int dk = dt * 16 + l16;
      const v8u lo = *reinterpret_cast<const v8u*>(&Vs[dk][half * 8]);
      const v8u hi = *reinterpret_cast<const v8u*>(&Vs[dk][half * 8 + 16]);
      oacc[dt] = wmma_bf16(pa, cat_bf(lo, hi), oacc[dt]);
    }
    __syncthreads();
  }

  // ---- normalize and store ----
  for (int dt = 0; dt < 4; ++dt) {
    const int dkc = h * DKc + dt * 16 + l16;
    for (int r = 0; r < 8; ++r) {
      const int m = b * Sc + q0 + r + half * 8;
      O[(size_t)m * Dc + dkc] = oacc[dt][r] / lrow[r];
    }
  }
}

// ---------------------------------------------------------------------------
// y = layernorm(x + r) * gamma + beta   (one row of D=512 per block)
// ---------------------------------------------------------------------------
__launch_bounds__(256) __global__
void add_layernorm_kernel(const float* __restrict__ X, const float* __restrict__ R,
                          const float* __restrict__ gamma,
                          const float* __restrict__ beta, float* __restrict__ Y) {
  __shared__ float red[16];
  const int row = blockIdx.x;
  const int tid = threadIdx.x;
  const float t0 = X[(size_t)row * Dc + tid] + R[(size_t)row * Dc + tid];
  const float t1 = X[(size_t)row * Dc + tid + 256] + R[(size_t)row * Dc + tid + 256];
  float s = t0 + t1, s2 = t0 * t0 + t1 * t1;
  for (int m = 1; m < 32; m <<= 1) {
    s += __shfl_xor(s, m);
    s2 += __shfl_xor(s2, m);
  }
  const int wid = tid >> 5, lane = tid & 31;
  if (lane == 0) { red[wid] = s; red[8 + wid] = s2; }
  __syncthreads();
  if (tid < 32) {
    float a = (tid < 8) ? red[tid] : 0.f;
    float c = (tid < 8) ? red[8 + tid] : 0.f;
    for (int m = 1; m < 8; m <<= 1) {
      a += __shfl_xor(a, m);
      c += __shfl_xor(c, m);
    }
    if (tid == 0) { red[0] = a; red[1] = c; }
  }
  __syncthreads();
  const float mean = red[0] * (1.f / Dc);
  const float var = red[1] * (1.f / Dc) - mean * mean;
  const float inv = rsqrtf(var + 1e-6f);
  Y[(size_t)row * Dc + tid] = (t0 - mean) * inv * gamma[tid] + beta[tid];
  Y[(size_t)row * Dc + tid + 256] =
      (t1 - mean) * inv * gamma[tid + 256] + beta[tid + 256];
}

// ---------------------------------------------------------------------------
// out[rs, :] = emb[seq[rs]] * sqrt(D) * (seq!=0) + PE[pos]
// ---------------------------------------------------------------------------
__launch_bounds__(256) __global__
void embed_kernel(const int* __restrict__ seq, const float* __restrict__ emb,
                  float* __restrict__ out) {
  const int rs = blockIdx.x;  // b*S + s
  const int pos = rs % Sc;
  const int tid = threadIdx.x;
  const int tok = seq[rs];
  const float scale = (tok != 0) ? 22.62741699796952f : 0.f;  // sqrt(512)
  for (int c = tid; c < Dc; c += 256) {
    const float div = __expf(-9.210340371976184f * (float)(c & ~1) / (float)Dc);
    const float ang = (float)pos * div;
    const float pe = (c & 1) ? __cosf(ang) : __sinf(ang);
    out[(size_t)rs * Dc + c] = emb[(size_t)tok * Dc + c] * scale + pe;
  }
}

// ---------------------------------------------------------------------------
// Host-side orchestration.
// ---------------------------------------------------------------------------
extern "C" void kernel_launch(void* const* d_in, const int* in_sizes, int n_in,
                              void* d_out, int out_size, void* d_ws,
                              size_t ws_size, hipStream_t stream) {
  (void)in_sizes; (void)n_in; (void)out_size; (void)ws_size;
  const int* src_seq = (const int*)d_in[0];
  const int* tgt_seq = (const int*)d_in[1];
  const float* emb = (const float*)d_in[2];
  auto P = [&](int i) { return (const float*)d_in[i]; };
  // encoder params (insertion order of setup_inputs dicts)
  const float *eWq = P(3), *eWk = P(4), *eWv = P(5), *eWo = P(6);
  const float *ebq = P(7), *ebk = P(8), *ebv = P(9), *ebo = P(10);
  const float *eW1 = P(11), *eb1 = P(12), *eW2 = P(13), *eb2 = P(14);
  const float *eln1s = P(15), *eln1b = P(16), *eln2s = P(17), *eln2b = P(18);
  // decoder params
  const float *dWq = P(19), *dWk = P(20), *dWv = P(21), *dWo = P(22);
  const float *dbq = P(23), *dbk = P(24), *dbv = P(25), *dbo = P(26);
  const float *cWq = P(27), *cWk = P(28), *cWv = P(29), *cWo = P(30);
  const float *cbq = P(31), *cbk = P(32), *cbv = P(33), *cbo = P(34);
  const float *dW1 = P(35), *db1 = P(36), *dW2 = P(37), *db2 = P(38);
  const float *dln1s = P(39), *dln1b = P(40), *dln2s = P(41), *dln2b = P(42);
  const float *dln3s = P(43), *dln3b = P(44);
  const float* out_b = P(45);

  // workspace layout
  float* ws = (float*)d_ws;
  float* x = ws;                   // encoder state / memory  [2048,512] f32
  float* y = x + (size_t)MR * Dc;  // decoder state           [2048,512] f32
  float* ao = y + (size_t)MR * Dc;
  float* tm = ao + (size_t)MR * Dc;
  float* ff = tm + (size_t)MR * Dc;  // [2048,2048] f32
  unsigned short* qh = (unsigned short*)(ff + (size_t)MR * Fc);  // bf16 q/k/v
  unsigned short* kh = qh + (size_t)MR * Dc;
  unsigned short* vh = kh + (size_t)MR * Dc;

  const dim3 blk(256);
  auto gemm = [&](const float* A, const float* W, const float* bias, void* C,
                  int M, int N, int K, int mode, int bt) {
    dim3 g(N / 128, M / 128);
    if (bt)
      gemm_bias_kernel<0, 1><<<g, blk, 0, stream>>>(A, W, bias, C, M, N, K);
    else if (mode == 2)
      gemm_bias_kernel<2, 0><<<g, blk, 0, stream>>>(A, W, bias, C, M, N, K);
    else if (mode == 1)
      gemm_bias_kernel<1, 0><<<g, blk, 0, stream>>>(A, W, bias, C, M, N, K);
    else
      gemm_bias_kernel<0, 0><<<g, blk, 0, stream>>>(A, W, bias, C, M, N, K);
  };
  auto attn = [&](const int* seq, int causal, float* o) {
    attention_kernel<<<dim3(Bc * Hc, Sc / 128), blk, 0, stream>>>(
        qh, kh, vh, seq, o, Sc, causal);
  };
  auto lnorm = [&](const float* X, const float* R, const float* g,
                   const float* bta, float* Y) {
    add_layernorm_kernel<<<MR, blk, 0, stream>>>(X, R, g, bta, Y);
  };

  // ---- encoder ----
  embed_kernel<<<MR, blk, 0, stream>>>(src_seq, emb, x);
  for (int l = 0; l < Lc; ++l) {
    const size_t wo = (size_t)l * Dc * Dc, bo_ = (size_t)l * Dc;
    gemm(x, eWq + wo, ebq + bo_, qh, MR, Dc, Dc, 2, 0);
    gemm(x, eWk + wo, ebk + bo_, kh, MR, Dc, Dc, 2, 0);
    gemm(x, eWv + wo, ebv + bo_, vh, MR, Dc, Dc, 2, 0);
    attn(src_seq, 0, ao);
    gemm(ao, eWo + wo, ebo + bo_, tm, MR, Dc, Dc, 0, 0);
    lnorm(x, tm, eln1s + bo_, eln1b + bo_, x);
    gemm(x, eW1 + (size_t)l * Dc * Fc, eb1 + (size_t)l * Fc, ff, MR, Fc, Dc, 1, 0);
    gemm(ff, eW2 + (size_t)l * Fc * Dc, eb2 + bo_, tm, MR, Dc, Fc, 0, 0);
    lnorm(x, tm, eln2s + bo_, eln2b + bo_, x);
  }

  // ---- decoder ----
  embed_kernel<<<MR, blk, 0, stream>>>(tgt_seq, emb, y);
  for (int l = 0; l < Lc; ++l) {
    const size_t wo = (size_t)l * Dc * Dc, bo_ = (size_t)l * Dc;
    // masked self-attention
    gemm(y, dWq + wo, dbq + bo_, qh, MR, Dc, Dc, 2, 0);
    gemm(y, dWk + wo, dbk + bo_, kh, MR, Dc, Dc, 2, 0);
    gemm(y, dWv + wo, dbv + bo_, vh, MR, Dc, Dc, 2, 0);
    attn(tgt_seq, 1, ao);
    gemm(ao, dWo + wo, dbo + bo_, tm, MR, Dc, Dc, 0, 0);
    lnorm(y, tm, dln1s + bo_, dln1b + bo_, y);
    // cross-attention against encoder memory x
    gemm(y, cWq + wo, cbq + bo_, qh, MR, Dc, Dc, 2, 0);
    gemm(x, cWk + wo, cbk + bo_, kh, MR, Dc, Dc, 2, 0);
    gemm(x, cWv + wo, cbv + bo_, vh, MR, Dc, Dc, 2, 0);
    attn(src_seq, 0, ao);
    gemm(ao, cWo + wo, cbo + bo_, tm, MR, Dc, Dc, 0, 0);
    lnorm(y, tm, dln2s + bo_, dln2b + bo_, y);
    // FFN
    gemm(y, dW1 + (size_t)l * Dc * Fc, db1 + (size_t)l * Fc, ff, MR, Fc, Dc, 1, 0);
    gemm(ff, dW2 + (size_t)l * Fc * Dc, db2 + bo_, tm, MR, Dc, Fc, 0, 0);
    lnorm(y, tm, dln3s + bo_, dln3b + bo_, y);
  }

  // ---- tied output projection: logits = y @ emb^T + out_b ----
  gemm(y, emb, out_b, d_out, MR, Vc, Dc, 0, 1);
}
